// FusedTriaffine_48576080118493
// MI455X (gfx1250) — compile-verified, matrix operand order checked by reading, not run on hardware
//
#include <hip/hip_runtime.h>
#include <hip/hip_bf16.h>
#include <math.h>

// ---------------------------------------------------------------------------
// Problem constants (from reference): B=2, L=160, D=768, H=128, R=4
// Padded size for (H+1)=129 dims -> 160 (multiple of 32 for WMMA K-steps).
// ---------------------------------------------------------------------------
#define BB   2
#define LL   160
#define DD   768
#define HH   128
#define RR   4
#define HP   160            // padded 129 -> 160
#define NKJ  (HH * HP)      // 20480 : flattened (k, j) / (q, m) index
#define NEG_SLOPE 0.1f

typedef __attribute__((ext_vector_type(16))) __bf16 v16bf;
typedef __attribute__((ext_vector_type(8)))  float  v8f;

union TileU { v16bf v; uint4 q[2]; };

// ---------------------------------------------------------------------------
// WMMA operand loaders per CDNA5 ISA 7.12.2 (wave32):
//   16-bit A 16x32 (MxK): lane = m + 16*hi ; element e holds
//       k = e + 8*hi + (e>=8 ? 8 : 0)
//   => per lane: two contiguous 8-half (16B) chunks at k0 = 8*hi and 16+8*hi.
//   B operands are stored transposed (N-major, [N x K] row-major) so the
//   identical load pattern applies with m -> n.
//   C/D f32 16x16: vgpr v holds row m = v + 8*hi, col n = lane&15.
// ---------------------------------------------------------------------------
__device__ inline v16bf load_tile_bf16(const __bf16* p, int ld) {
  const int lane = threadIdx.x & 31;
  const int row  = lane & 15;
  const int hi   = lane >> 4;
  const __bf16* q = p + row * ld + hi * 8;
  TileU t;
  t.q[0] = *reinterpret_cast<const uint4*>(q);        // k = 8*hi + [0..7]
  t.q[1] = *reinterpret_cast<const uint4*>(q + 16);   // k = 16 + 8*hi + [0..7]
  return t.v;
}

__device__ inline v8f wmma_op(v16bf a, v16bf b, v8f c) {
  return __builtin_amdgcn_wmma_f32_16x16x32_bf16(
      /*neg_a=*/false, a, /*neg_b=*/false, b,
      /*c_mod=*/(short)0, c, /*reuse_a=*/false, /*reuse_b=*/false);
}

// Single 16x16 tile accumulate (used by small projection GEMMs)
__device__ inline v8f wmma_accum(const __bf16* A, int lda,
                                 const __bf16* BT, int ldb,
                                 int ksteps, v8f acc) {
  for (int s = 0; s < ksteps; ++s) {
    v16bf a = load_tile_bf16(A + s * 32, lda);
    v16bf b = load_tile_bf16(BT + s * 32, ldb);
    acc = wmma_op(a, b, acc);
  }
  return acc;
}

// 2x2 register-blocked 32x32 output block: 4 fragment loads -> 4 WMMAs
// (1 KB of operands per WMMA instead of 2 KB -> halves L2/HBM traffic)
__device__ inline void wmma_block_2x2(const __bf16* A, int lda,
                                      const __bf16* BT, int ldb,
                                      int ksteps, v8f acc[2][2]) {
  for (int s = 0; s < ksteps; ++s) {
    v16bf a0 = load_tile_bf16(A + s * 32, lda);
    v16bf a1 = load_tile_bf16(A + 16 * lda + s * 32, lda);
    v16bf b0 = load_tile_bf16(BT + s * 32, ldb);
    v16bf b1 = load_tile_bf16(BT + 16 * ldb + s * 32, ldb);
    acc[0][0] = wmma_op(a0, b0, acc[0][0]);
    acc[0][1] = wmma_op(a0, b1, acc[0][1]);
    acc[1][0] = wmma_op(a1, b0, acc[1][0]);
    acc[1][1] = wmma_op(a1, b1, acc[1][1]);
  }
}

// 1x2 N-blocked: one A fragment feeds two WMMAs (A usually LDS-resident)
__device__ inline void wmma_block_1x2(const __bf16* A, int lda,
                                      const __bf16* BT, int ldb,
                                      int ksteps, v8f* acc0, v8f* acc1) {
  for (int s = 0; s < ksteps; ++s) {
    v16bf a  = load_tile_bf16(A + s * 32, lda);
    v16bf b0 = load_tile_bf16(BT + s * 32, ldb);
    v16bf b1 = load_tile_bf16(BT + 16 * ldb + s * 32, ldb);
    *acc0 = wmma_op(a, b0, *acc0);
    *acc1 = wmma_op(a, b1, *acc1);
  }
}

__device__ inline void store_tile_f32(float* p, int ld, v8f acc) {
  const int lane = threadIdx.x & 31;
  const int n = lane & 15, hi = lane >> 4;
#pragma unroll
  for (int v = 0; v < 8; ++v) p[(v + hi * 8) * ld + n] = acc[v];
}

__device__ inline void store_tile_bf16(__bf16* p, int ld, v8f acc) {
  const int lane = threadIdx.x & 31;
  const int n = lane & 15, hi = lane >> 4;
#pragma unroll
  for (int v = 0; v < 8; ++v) p[(v + hi * 8) * ld + n] = (__bf16)acc[v];
}

__device__ inline float leaky(float v) { return v >= 0.f ? v : NEG_SLOPE * v; }

// ---------------------------------------------------------------------------
// Precision-prep kernels: f32 -> bf16 casts / transposes / permutes
// ---------------------------------------------------------------------------
__global__ void k_cast(const float* in, __bf16* out, int n) {
  for (int i = blockIdx.x * blockDim.x + threadIdx.x; i < n; i += gridDim.x * blockDim.x)
    out[i] = (__bf16)in[i];
}

// out[n*K + k] = in[k*N + n]
__global__ void k_transpose_cast(const float* in, __bf16* out, int N, int K) {
  int total = N * K;
  for (int i = blockIdx.x * blockDim.x + threadIdx.x; i < total; i += gridDim.x * blockDim.x) {
    int k = i % K, n = i / K;
    out[i] = (__bf16)in[k * N + n];
  }
}

// W_tri [R,129,128,129] -> WtriT[((r*H + k)*HP + j)*HP + i], zero-padded
__global__ void k_wtri(const float* in, __bf16* out) {
  int total = RR * HH * HP * HP;
  for (int idx = blockIdx.x * blockDim.x + threadIdx.x; idx < total; idx += gridDim.x * blockDim.x) {
    int ii = idx % HP;
    int jj = (idx / HP) % HP;
    int kk = (idx / (HP * HP)) % HH;
    int r  = idx / (HP * HP * HH);
    float v = 0.f;
    if (ii < HH + 1 && jj < HH + 1)
      v = in[((r * (HH + 1) + ii) * HH + kk) * (HH + 1) + jj];
    out[idx] = (__bf16)v;
  }
}

// W_final [129,128,129] -> WfinT[(q*HP + m)*HP + n], zero-padded
__global__ void k_wfin(const float* in, __bf16* out) {
  int total = HH * HP * HP;
  for (int idx = blockIdx.x * blockDim.x + threadIdx.x; idx < total; idx += gridDim.x * blockDim.x) {
    int n = idx % HP;
    int m = (idx / HP) % HP;
    int q = idx / (HP * HP);
    float v = 0.f;
    if (n < HH + 1 && m < HH + 1)
      v = in[(n * HH + q) * (HH + 1) + m];
    out[idx] = (__bf16)v;
  }
}

// Fill pad columns [H..HP) of the four padded activations; col H gets 1.0
__global__ void k_pad(__bf16* h_tail_p, __bf16* h_in_p, __bf16* x_head_p, __bf16* x_tail_p) {
  int total = 4 * BB * LL * (HP - HH);
  for (int idx = blockIdx.x * blockDim.x + threadIdx.x; idx < total; idx += gridDim.x * blockDim.x) {
    int c   = HH + (idx % (HP - HH));
    int row = (idx / (HP - HH)) % (BB * LL);
    int a   = idx / ((HP - HH) * BB * LL);
    __bf16 v = (__bf16)((c == HH) ? 1.0f : 0.0f);
    __bf16* p = (a == 0) ? h_tail_p : (a == 1) ? h_in_p : (a == 2) ? x_head_p : x_tail_p;
    p[row * HP + c] = v;
  }
}

// ---------------------------------------------------------------------------
// Kernel A1: hidden = leaky(x @ W_span + b_span), scatter 4 chunks (bf16)
//   M = B*L = 320 (20 tiles), N = 4H = 512 (32 tiles), K = 768 (24 steps)
// ---------------------------------------------------------------------------
__global__ void __launch_bounds__(32)
k_hidden(const __bf16* xb, const __bf16* WT, const float* bias,
         __bf16* h_reprT, __bf16* h_head, __bf16* h_tail_p, __bf16* h_in_p) {
  int id = blockIdx.x;
  int nt = id % 32, mt = id / 32;
  int m0 = mt * 16, n0 = nt * 16;
  v8f acc = {};
  acc = wmma_accum(xb + m0 * DD, DD, WT + n0 * DD, DD, DD / 32, acc);

  const int lane = threadIdx.x & 31, nl = lane & 15, hi = lane >> 4;
  const int col = n0 + nl;
  const float bc = bias[col];
  const int c = col & (HH - 1);
#pragma unroll
  for (int v = 0; v < 8; ++v) {
    int row = m0 + v + hi * 8;            // 0..319
    int b = row / LL, l = row % LL;
    __bf16 bv = (__bf16)leaky(acc[v] + bc);
    switch (col >> 7) {
      case 0:  h_reprT [(b * HH + c) * HP + l] = bv; break;  // h_in_repr, [b,h,z]
      case 1:  h_head  [(b * LL + l) * HH + c] = bv; break;  // [b,l,h]
      case 2:  h_tail_p[(b * LL + l) * HP + c] = bv; break;  // [b,l,HP]
      default: h_in_p  [(b * LL + l) * HP + c] = bv; break;  // [b,l,HP]
    }
  }
}

// Kernel A2: scoring = leaky(x @ W_score + b_score), scatter 2 padded chunks
__global__ void __launch_bounds__(32)
k_scoring(const __bf16* xb, const __bf16* WT, const float* bias,
          __bf16* x_head_p, __bf16* x_tail_p) {
  int id = blockIdx.x;
  int nt = id % 16, mt = id / 16;
  int m0 = mt * 16, n0 = nt * 16;
  v8f acc = {};
  acc = wmma_accum(xb + m0 * DD, DD, WT + n0 * DD, DD, DD / 32, acc);

  const int lane = threadIdx.x & 31, nl = lane & 15, hi = lane >> 4;
  const int col = n0 + nl;
  const float bc = bias[col];
  const int c = col & (HH - 1);
#pragma unroll
  for (int v = 0; v < 8; ++v) {
    int row = m0 + v + hi * 8;
    int b = row / LL, l = row % LL;
    __bf16 bv = (__bf16)leaky(acc[v] + bc);
    if (col < HH) x_head_p[(b * LL + l) * HP + c] = bv;
    else          x_tail_p[(b * LL + l) * HP + c] = bv;
  }
}

// ---------------------------------------------------------------------------
// Kernel B: T1[b,r,y,(k,j)] = sum_i h_tail_p[b,y,i] * W_tri[r,i,k,j]
// 2x2 register-blocked (32x32 per wave); stored transposed per (b,r,y) as
// T1T[j (HP rows) x k (H cols)] bf16.
//   grid: b * r * 5 yblocks * 640 nblocks
// ---------------------------------------------------------------------------
__global__ void __launch_bounds__(32)
k_t1(const __bf16* h_tail_p, const __bf16* WtriT, __bf16* T1T) {
  int id = blockIdx.x;
  int nb = id % (NKJ / 32);
  int yb = (id / (NKJ / 32)) % (LL / 32);
  int r  = (id / (NKJ / 32 * LL / 32)) % RR;
  int b  =  id / (NKJ / 32 * LL / 32 * RR);
  int y0 = yb * 32, n0 = nb * 32;

  v8f acc[2][2] = {};
  wmma_block_2x2(h_tail_p + (b * LL + y0) * HP, HP,
                 WtriT + (r * NKJ + n0) * HP, HP, HP / 32, acc);

  const int lane = threadIdx.x & 31, nl = lane & 15, hi = lane >> 4;
#pragma unroll
  for (int mi = 0; mi < 2; ++mi) {
#pragma unroll
    for (int ni = 0; ni < 2; ++ni) {
      const int n = n0 + ni * 16 + nl;
      const int kk = n / HP, j = n % HP;
#pragma unroll
      for (int v = 0; v < 8; ++v) {
        int y = y0 + mi * 16 + v + hi * 8;
        T1T[((b * RR + r) * LL + y) * NKJ + j * HH + kk] = (__bf16)acc[mi][ni][v];
      }
    }
  }
}

// ---------------------------------------------------------------------------
// Kernel D1: Afin[b,x,(q,m)] = sum_n x_head_p[b,x,n] * W_final[n,q,m]  (bf16)
//   2x2 blocked: M = 320 (10 blocks), N = NKJ (640 blocks), K = HP (5 steps)
// ---------------------------------------------------------------------------
__global__ void __launch_bounds__(32)
k_afin(const __bf16* x_head_p, const __bf16* WfinT, __bf16* Abf) {
  int id = blockIdx.x;
  int nb = id % (NKJ / 32), mb = id / (NKJ / 32);
  int m0 = mb * 32, n0 = nb * 32;
  v8f acc[2][2] = {};
  wmma_block_2x2(x_head_p + m0 * HP, HP, WfinT + n0 * HP, HP, HP / 32, acc);
#pragma unroll
  for (int mi = 0; mi < 2; ++mi)
#pragma unroll
    for (int ni = 0; ni < 2; ++ni)
      store_tile_bf16(Abf + (m0 + mi * 16) * NKJ + n0 + ni * 16, NKJ, acc[mi][ni]);
}

// Kernel D2: V[b,x,y,q] = sum_m x_tail_p[b,y,m] * Afin[b,x,q,m]  (f32)
//   2x2 blocked: grid = b * x * 5 yblocks * 4 qblocks
__global__ void __launch_bounds__(32)
k_v(const __bf16* x_tail_p, const __bf16* Abf, float* V) {
  int id = blockIdx.x;
  int qb = id % (HH / 32);
  int yb = (id / (HH / 32)) % (LL / 32);
  int x  = (id / (HH / 32 * LL / 32)) % LL;
  int b  =  id / (HH / 32 * LL / 32 * LL);
  int y0 = yb * 32, q0 = qb * 32;
  v8f acc[2][2] = {};
  wmma_block_2x2(x_tail_p + (b * LL + y0) * HP, HP,
                 Abf + (b * LL + x) * NKJ + q0 * HP, HP, HP / 32, acc);
#pragma unroll
  for (int mi = 0; mi < 2; ++mi)
#pragma unroll
    for (int ni = 0; ni < 2; ++ni)
      store_tile_f32(V + ((b * LL + x) * LL + y0 + mi * 16) * HH + q0 + ni * 16,
                     HH, acc[mi][ni]);
}

// ---------------------------------------------------------------------------
// Kernel C (fused core): per workgroup = (b, r, y, x-tile of 16)
//   T2 = h_head_tile @ T1T[b,r,y]           -> LDS bf16 [16 x HP]
//   logits = T2 @ h_in_p^T                  -> LDS f32  [16 x L]
//   softmax over z                          -> LDS bf16 alpha [16 x L]
//   span = alpha @ h_in_repr                -> LDS bf16 [16 x H]
//   spanq = leaky(span @ W_spanmlp + bias)  -> LDS f32  [16 x H]
//   score[x] = sum_q spanq[x,q] * V[b,x,y,q]
// All GEMM stages 1x2 N-blocked: each LDS A-fragment feeds two WMMAs.
// ---------------------------------------------------------------------------
__global__ void __launch_bounds__(32)
k_fused(const __bf16* h_head, const __bf16* T1T, const __bf16* h_in_p,
        const __bf16* h_in_reprT, const __bf16* WmlpT, const float* b_mlp,
        const float* V, float* out) {
  __shared__ __align__(16) __bf16 s_t2[16 * HP];
  __shared__ __align__(16) float  s_log[16 * LL];
  __shared__ __align__(16) __bf16 s_alpha[16 * LL];
  __shared__ __align__(16) __bf16 s_span[16 * HH];
  __shared__ __align__(16) float  s_spanq[16 * HH];

  int id = blockIdx.x;
  int xt = id % (LL / 16);
  int y  = (id / (LL / 16)) % LL;
  int r  = (id / (LL / 16 * LL)) % RR;
  int b  =  id / (LL / 16 * LL * RR);
  const int x0 = xt * 16;
  const int lane = threadIdx.x & 31, nl = lane & 15, hi = lane >> 4;

  // --- T2[x, j] = sum_k h_head[x,k] * T1[y,k,j] ---
  const __bf16* Ah  = h_head + (b * LL + x0) * HH;
  const __bf16* BT1 = T1T + (size_t)((b * RR + r) * LL + y) * NKJ;
  for (int jt = 0; jt < HP / 32; ++jt) {
    v8f a0 = {}, a1 = {};
    wmma_block_1x2(Ah, HH, BT1 + jt * 32 * HH, HH, HH / 32, &a0, &a1);
#pragma unroll
    for (int v = 0; v < 8; ++v) {
      s_t2[(v + hi * 8) * HP + jt * 32 + nl]      = (__bf16)a0[v];
      s_t2[(v + hi * 8) * HP + jt * 32 + 16 + nl] = (__bf16)a1[v];
    }
  }
  __syncthreads();

  // --- logits[x, z] = sum_j T2[x,j] * h_in_p[z,j] ---
  const __bf16* Bin = h_in_p + b * LL * HP;
  for (int zt = 0; zt < LL / 32; ++zt) {
    v8f a0 = {}, a1 = {};
    wmma_block_1x2(s_t2, HP, Bin + zt * 32 * HP, HP, HP / 32, &a0, &a1);
#pragma unroll
    for (int v = 0; v < 8; ++v) {
      s_log[(v + hi * 8) * LL + zt * 32 + nl]      = a0[v];
      s_log[(v + hi * 8) * LL + zt * 32 + 16 + nl] = a1[v];
    }
  }
  __syncthreads();

  // --- softmax over z (lane = local x row) ---
  if (lane < 16) {
    float mx = -1e30f;
    for (int z = 0; z < LL; ++z) mx = fmaxf(mx, s_log[lane * LL + z]);
    float sum = 0.f;
    for (int z = 0; z < LL; ++z) {
      float e = __expf(s_log[lane * LL + z] - mx);
      s_log[lane * LL + z] = e;
      sum += e;
    }
    float inv = 1.f / sum;
    for (int z = 0; z < LL; ++z)
      s_alpha[lane * LL + z] = (__bf16)(s_log[lane * LL + z] * inv);
  }
  __syncthreads();

  // --- span[x, h] = sum_z alpha[x,z] * h_in_repr[z,h] ---
  const __bf16* Brep = h_in_reprT + b * HH * HP;   // [h rows][z cols], ld = HP
  for (int ht = 0; ht < HH / 32; ++ht) {
    v8f a0 = {}, a1 = {};
    wmma_block_1x2(s_alpha, LL, Brep + ht * 32 * HP, HP, LL / 32, &a0, &a1);
#pragma unroll
    for (int v = 0; v < 8; ++v) {
      s_span[(v + hi * 8) * HH + ht * 32 + nl]      = (__bf16)a0[v];
      s_span[(v + hi * 8) * HH + ht * 32 + 16 + nl] = (__bf16)a1[v];
    }
  }
  __syncthreads();

  // --- spanq = leaky(span @ W_spanmlp + bias) ---
  for (int qt = 0; qt < HH / 32; ++qt) {
    v8f a0 = {}, a1 = {};
    wmma_block_1x2(s_span, HH, WmlpT + qt * 32 * HH, HH, HH / 32, &a0, &a1);
    float bq0 = b_mlp[qt * 32 + nl];
    float bq1 = b_mlp[qt * 32 + 16 + nl];
#pragma unroll
    for (int v = 0; v < 8; ++v) {
      s_spanq[(v + hi * 8) * HH + qt * 32 + nl]      = leaky(a0[v] + bq0);
      s_spanq[(v + hi * 8) * HH + qt * 32 + 16 + nl] = leaky(a1[v] + bq1);
    }
  }
  __syncthreads();

  // --- score[b,r,x,y] = sum_q spanq[x,q] * V[b,x,y,q] ---
  if (lane < 16) {
    int xg = x0 + lane;
    const float4* vp = reinterpret_cast<const float4*>(
        V + (size_t)((b * LL + xg) * LL + y) * HH);
    const float4* sp = reinterpret_cast<const float4*>(s_spanq + lane * HH);
    float ssum = 0.f;
#pragma unroll 4
    for (int q4 = 0; q4 < HH / 4; ++q4) {
      float4 vv = vp[q4];
      float4 ss = sp[q4];
      ssum += ss.x * vv.x + ss.y * vv.y + ss.z * vv.z + ss.w * vv.w;
    }
    out[((size_t)(b * RR + r) * LL + xg) * LL + y] = ssum;
  }
}

// ---------------------------------------------------------------------------
// Host-side launch
// ---------------------------------------------------------------------------
extern "C" void kernel_launch(void* const* d_in, const int* in_sizes, int n_in,
                              void* d_out, int out_size, void* d_ws, size_t ws_size,
                              hipStream_t stream) {
  const float* x        = (const float*)d_in[0];
  const float* W_span   = (const float*)d_in[1];
  const float* b_span   = (const float*)d_in[2];
  const float* W_score  = (const float*)d_in[3];
  const float* b_score  = (const float*)d_in[4];
  const float* W_tri    = (const float*)d_in[5];
  const float* W_mlp    = (const float*)d_in[6];
  const float* b_mlp    = (const float*)d_in[7];
  const float* W_final  = (const float*)d_in[8];
  float* out = (float*)d_out;

  // workspace partition (all 256B-aligned); total ~127 MB
  size_t off = 0;
  auto alloc = [&](size_t bytes) {
    off = (off + 255) & ~(size_t)255;
    void* p = (char*)d_ws + off;
    off += bytes;
    return p;
  };
  __bf16* xb       = (__bf16*)alloc((size_t)BB * LL * DD * 2);
  __bf16* WspanT   = (__bf16*)alloc((size_t)4 * HH * DD * 2);
  __bf16* WscoreT  = (__bf16*)alloc((size_t)2 * HH * DD * 2);
  __bf16* WtriT    = (__bf16*)alloc((size_t)RR * NKJ * HP * 2);
  __bf16* WfinT    = (__bf16*)alloc((size_t)NKJ * HP * 2);
  __bf16* WmlpT    = (__bf16*)alloc((size_t)HH * HH * 2);
  __bf16* h_head   = (__bf16*)alloc((size_t)BB * LL * HH * 2);
  __bf16* h_tail_p = (__bf16*)alloc((size_t)BB * LL * HP * 2);
  __bf16* h_in_p   = (__bf16*)alloc((size_t)BB * LL * HP * 2);
  __bf16* h_reprT  = (__bf16*)alloc((size_t)BB * HH * HP * 2);
  __bf16* x_head_p = (__bf16*)alloc((size_t)BB * LL * HP * 2);
  __bf16* x_tail_p = (__bf16*)alloc((size_t)BB * LL * HP * 2);
  __bf16* T1T      = (__bf16*)alloc((size_t)BB * RR * LL * NKJ * 2);
  __bf16* Abf      = (__bf16*)alloc((size_t)BB * LL * NKJ * 2);
  float*  V        = (float*) alloc((size_t)BB * LL * LL * HH * 4);
  (void)ws_size; (void)n_in; (void)in_sizes; (void)out_size;

  const int T = 256;
  auto gs = [](int n, int t) { return (n + t - 1) / t; };

  // precision prep
  k_cast<<<gs(BB * LL * DD, T), T, 0, stream>>>(x, xb, BB * LL * DD);
  k_transpose_cast<<<gs(4 * HH * DD, T), T, 0, stream>>>(W_span, WspanT, 4 * HH, DD);
  k_transpose_cast<<<gs(2 * HH * DD, T), T, 0, stream>>>(W_score, WscoreT, 2 * HH, DD);
  k_transpose_cast<<<gs(HH * HH, T), T, 0, stream>>>(W_mlp, WmlpT, HH, HH);
  k_wtri<<<gs(RR * HH * HP * HP, T), T, 0, stream>>>(W_tri, WtriT);
  k_wfin<<<gs(HH * HP * HP, T), T, 0, stream>>>(W_final, WfinT);
  k_pad<<<gs(4 * BB * LL * (HP - HH), T), T, 0, stream>>>(h_tail_p, h_in_p, x_head_p, x_tail_p);

  // projections (WMMA GEMMs, 1 wave / 16x16 tile)
  k_hidden<<<(BB * LL / 16) * (4 * HH / 16), 32, 0, stream>>>(
      xb, WspanT, b_span, h_reprT, h_head, h_tail_p, h_in_p);
  k_scoring<<<(BB * LL / 16) * (2 * HH / 16), 32, 0, stream>>>(
      xb, WscoreT, b_score, x_head_p, x_tail_p);

  // T1 = h_tail_p x W_tri (2x2 blocked; stored transposed per (b,r,y))
  k_t1<<<BB * RR * (LL / 32) * (NKJ / 32), 32, 0, stream>>>(h_tail_p, WtriT, T1T);

  // final biaffine precompute: Afin then V (both 2x2 blocked)
  k_afin<<<(BB * LL / 32) * (NKJ / 32), 32, 0, stream>>>(x_head_p, WfinT, Abf);
  k_v<<<BB * LL * (LL / 32) * (HH / 32), 32, 0, stream>>>(x_tail_p, Abf, V);

  // fused T2 -> logits -> softmax -> span -> span-MLP -> score
  k_fused<<<BB * RR * LL * (LL / 16), 32, 0, stream>>>(
      h_head, T1T, h_in_p, h_reprT, WmlpT, b_mlp, V, out);
}
